// DeepST_model_79422535237670
// MI455X (gfx1250) — matrix operand order repaired and checked by reading, not compiled, not run on hardware
//
#include <hip/hip_runtime.h>
#include <math.h>

#define Nn   50000
#define DIN  3000
#define Ee   300000

// ---------------- WMMA types ----------------
typedef __attribute__((ext_vector_type(8)))  float        v8f;
typedef __attribute__((ext_vector_type(16))) __bf16       v16bf;
typedef __attribute__((ext_vector_type(8)))  unsigned int v8u;

union BFV { v8u u; v16bf b; };

__device__ __forceinline__ unsigned int pack_bf2(float f0, float f1) {
    unsigned int u0 = __float_as_uint(f0); u0 += 0x7FFFu + ((u0 >> 16) & 1u);
    unsigned int u1 = __float_as_uint(f1); u1 += 0x7FFFu + ((u1 >> 16) & 1u);
    return (u0 >> 16) | (u1 & 0xFFFF0000u);
}

__device__ __forceinline__ float eluf(float y) { return y > 0.f ? y : (expf(y) - 1.f); }

// ---------------- weight prep: f32 -> bf16, swizzled into contiguous-k B layout (k = 16*half + s) ----
// B vector (32Kx16N, bf16): lane l -> n = nb + (l&15); K half = l>>4; VGPR j packs K = 16*(l>>4)+2j, +1
// u32 slot index for (chunk, lane, j) = (chunk*32 + lane)*8 + j

__global__ void prep_w1(const float* __restrict__ W, unsigned int* __restrict__ wb) {
    int p = blockIdx.x * 256 + threadIdx.x;            // 188*256 = 48128 exactly
    int j = p & 7, lane = (p >> 3) & 31, chunk = p >> 8; // chunk = t*2 + h
    int t = chunk >> 1, h = chunk & 1;
    int n  = h * 16 + (lane & 15);
    int kl = ((lane >> 4) << 4) + (j << 1);
    int K  = t * 32 + kl;
    float f0 = (K     < DIN) ? W[(size_t)K * 32 + n]       : 0.f;
    float f1 = (K + 1 < DIN) ? W[(size_t)(K + 1) * 32 + n] : 0.f;
    wb[p] = pack_bf2(f0, f1);
}

__global__ void prep_wout(const float* __restrict__ W, unsigned int* __restrict__ wb) {
    int p = blockIdx.x * 256 + threadIdx.x;            // 188 n-chunks of 16 (N padded to 3008)
    int j = p & 7, lane = (p >> 3) & 31, c = p >> 8;
    int n  = c * 16 + (lane & 15);
    int kl = ((lane >> 4) << 4) + (j << 1);            // K in [0,32)
    float f0 = (n < DIN) ? W[(size_t)kl * DIN + n]       : 0.f;
    float f1 = (n < DIN) ? W[(size_t)(kl + 1) * DIN + n] : 0.f;
    wb[p] = pack_bf2(f0, f1);
}

// ---------------- GEMM 1: h1 = ELU(BN(x @ W1 + b1))  [50000,3000]x[3000,32] ------
__global__ void __launch_bounds__(256) gemm1_wmma(
    const float* __restrict__ x, const unsigned int* __restrict__ wb,
    const float* __restrict__ lb, const float* __restrict__ bg, const float* __restrict__ bb,
    float* __restrict__ h1)
{
    int wid = (int)blockIdx.x * 8 + ((int)threadIdx.x >> 5);
    if (wid >= (Nn / 16)) return;                       // uniform per-wave
    int lane = threadIdx.x & 31;
    int half = lane >> 4;
    int m0   = wid << 4;
    const float* xr = x + (size_t)(m0 + (lane & 15)) * DIN;
    v8f acc0 = {0.f,0.f,0.f,0.f,0.f,0.f,0.f,0.f};
    v8f acc1 = acc0;
    const v8u* bp = (const v8u*)wb + lane;

#pragma unroll 1
    for (int t = 0; t < 93; ++t) {
        int kb = t << 5;
        __builtin_prefetch(xr + kb + 256, 0, 1);        // global_prefetch_b8
        BFV a;
#pragma unroll
        for (int j = 0; j < 8; ++j) {
            int k = kb + (j << 1) + (half << 4);
            a.u[j] = pack_bf2(xr[k], xr[k + 1]);
        }
        BFV b0, b1; b0.u = bp[t * 64]; b1.u = bp[t * 64 + 32];
        acc0 = __builtin_amdgcn_wmma_f32_16x16x32_bf16(false, a.b, false, b0.b, (short)0, acc0, false, false);
        acc1 = __builtin_amdgcn_wmma_f32_16x16x32_bf16(false, a.b, false, b1.b, (short)0, acc1, false, false);
    }
    { // K tail: 2976..3007, valid < 3000 (pairs are even-aligned so both-or-neither valid)
        BFV a;
#pragma unroll
        for (int j = 0; j < 8; ++j) {
            int k = 2976 + (j << 1) + (half << 4);
            a.u[j] = (k < DIN) ? pack_bf2(xr[k], xr[k + 1]) : 0u;
        }
        BFV b0, b1; b0.u = bp[93 * 64]; b1.u = bp[93 * 64 + 32];
        acc0 = __builtin_amdgcn_wmma_f32_16x16x32_bf16(false, a.b, false, b0.b, (short)0, acc0, false, false);
        acc1 = __builtin_amdgcn_wmma_f32_16x16x32_bf16(false, a.b, false, b1.b, (short)0, acc1, false, false);
    }
    // epilogue: C layout: element (r, lane) -> M = r + 8*(lane>>4), N = lane&15 (+16 for acc1)
    float rs = rsqrtf(1.0f + 1e-3f);
    int nA = lane & 15, nB = nA + 16;
    float sA = bg[nA] * rs, sB = bg[nB] * rs;
    float cA = bb[nA], cB = bb[nB];
    float iA = lb[nA], iB = lb[nB];
#pragma unroll
    for (int r = 0; r < 8; ++r) {
        int row = m0 + r + (half << 3);
        float y0 = eluf((acc0[r] + iA) * sA + cA);
        float y1 = eluf((acc1[r] + iB) * sB + cB);
        h1[(size_t)row * 32 + nA] = y0;
        h1[(size_t)row * 32 + nB] = y1;
    }
}

// ---------------- GEMM 2: de_feat = dec_h @ W_out + b  [50000,32]x[32,3000] ------
__global__ void __launch_bounds__(256) gemm2_wmma(
    const float* __restrict__ dh, const unsigned int* __restrict__ wb,
    const float* __restrict__ bo, float* __restrict__ de)
{
    int wid = (int)blockIdx.x * 8 + ((int)threadIdx.x >> 5);
    if (wid >= (Nn / 16) * 47) return;                  // 47 n-groups of 64 cols (N padded to 3008)
    int mt = wid / 47, ng = wid % 47;
    int lane = threadIdx.x & 31, half = lane >> 4;
    int m0 = mt << 4;
    const float* xr = dh + (size_t)(m0 + (lane & 15)) * 32;
    BFV a;
#pragma unroll
    for (int j = 0; j < 8; ++j) {
        int k = (j << 1) + (half << 4);
        a.u[j] = pack_bf2(xr[k], xr[k + 1]);
    }
    v8f z = {0.f,0.f,0.f,0.f,0.f,0.f,0.f,0.f};
    v8f acc[4]; 
#pragma unroll
    for (int c = 0; c < 4; ++c) acc[c] = z;
    const v8u* bp = (const v8u*)wb + lane;
#pragma unroll
    for (int c = 0; c < 4; ++c) {
        BFV bv; bv.u = bp[(ng * 4 + c) * 32];
        acc[c] = __builtin_amdgcn_wmma_f32_16x16x32_bf16(false, a.b, false, bv.b, (short)0, acc[c], false, false);
    }
#pragma unroll
    for (int c = 0; c < 4; ++c) {
        int n = (ng * 4 + c) * 16 + (lane & 15);
        if (n < DIN) {
            float b = bo[n];
#pragma unroll
            for (int r = 0; r < 8; ++r) {
                int row = m0 + r + (half << 3);
                de[(size_t)row * DIN + n] = acc[c][r] + b;
            }
        }
    }
}

// ---------------- generic per-row linear, fully register-resident --------------
// MODE 0: none   MODE 1: BN(eps)+ELU   MODE 2: LayerNorm+ReLU
template<int K, int C, int MODE>
__global__ void __launch_bounds__(256) lin_k(
    const float* __restrict__ in, const float* __restrict__ W, const float* __restrict__ B,
    const float* __restrict__ G, const float* __restrict__ Bb, float eps,
    float* __restrict__ out, float* __restrict__ out2, int n)
{
    __shared__ float sW[K * C];
    __shared__ float sB[C], sG[C], sBb[C];
    for (int i = threadIdx.x; i < K * C; i += 256) sW[i] = W[i];
    for (int i = threadIdx.x; i < C; i += 256) {
        sB[i]  = B  ? B[i]  : 0.f;
        sG[i]  = G  ? G[i]  : 1.f;
        sBb[i] = Bb ? Bb[i] : 0.f;
    }
    __syncthreads();
    int row = blockIdx.x * 256 + threadIdx.x;
    if (row >= n) return;
    float a[K];
#pragma unroll
    for (int k = 0; k < K; ++k) a[k] = in[(size_t)row * K + k];
    float o[C];
#pragma unroll
    for (int c = 0; c < C; ++c) {
        float s = sB[c];
#pragma unroll
        for (int k = 0; k < K; ++k) s += a[k] * sW[k * C + c];
        o[c] = s;
    }
    if (MODE == 1) {
        float rs = rsqrtf(1.f + eps);
#pragma unroll
        for (int c = 0; c < C; ++c) o[c] = eluf(o[c] * (sG[c] * rs) + sBb[c]);
    } else if (MODE == 2) {
        float mu = 0.f;
#pragma unroll
        for (int c = 0; c < C; ++c) mu += o[c];
        mu *= (1.f / C);
        float var = 0.f;
#pragma unroll
        for (int c = 0; c < C; ++c) { float d = o[c] - mu; var += d * d; }
        var *= (1.f / C);
        float rs = rsqrtf(var + 1e-5f);
#pragma unroll
        for (int c = 0; c < C; ++c) {
            float y = (o[c] - mu) * rs * sG[c] + sBb[c];
            o[c] = y > 0.f ? y : 0.f;
        }
    }
#pragma unroll
    for (int c = 0; c < C; ++c) {
        out[(size_t)row * C + c] = o[c];
        if (out2) out2[(size_t)row * C + c] = o[c];
    }
}

// ---------------- transformer qkv + global reductions ---------------------------
// red: [0]=sum q^2, [1]=sum k^2, [2..65]=ksum, [66..4161]=KV(64x64)
__global__ void __launch_bounds__(256) qkv_kernel(
    const float* __restrict__ t,
    const float* __restrict__ Wq, const float* __restrict__ bq,
    const float* __restrict__ Wk, const float* __restrict__ bk,
    const float* __restrict__ Wv, const float* __restrict__ bv,
    float* __restrict__ qb, float* __restrict__ kbuf, float* __restrict__ vbuf,
    float* __restrict__ red, int n)
{
    __shared__ float ts[4096], ks[4096], vs[4096];
    __shared__ float s_ssq, s_ssk;
    int tid = threadIdx.x;
    int row0 = blockIdx.x * 64;
    for (int i = tid; i < 4096; i += 256) {
        int r = i >> 6, c = i & 63; int row = row0 + r;
        ts[i] = (row < n) ? t[(size_t)row * 64 + c] : 0.f;
    }
    if (tid == 0) { s_ssq = 0.f; s_ssk = 0.f; }
    __syncthreads();
    int r = tid >> 2, c0 = (tid & 3) * 16;
    int row = row0 + r;
    bool valid = row < n;
    float ssq = 0.f, ssk = 0.f;
    for (int cc = 0; cc < 16; ++cc) {
        int c = c0 + cc;
        float sq = bq[c], sk = bk[c], sv = bv[c];
#pragma unroll
        for (int k = 0; k < 64; ++k) {
            float a = ts[r * 64 + k];
            sq += a * Wq[k * 64 + c];
            sk += a * Wk[k * 64 + c];
            sv += a * Wv[k * 64 + c];
        }
        if (!valid) { sq = 0.f; sk = 0.f; sv = 0.f; }
        ks[r * 64 + c] = sk; vs[r * 64 + c] = sv;
        if (valid) {
            size_t o = (size_t)row * 64 + c;
            qb[o] = sq; kbuf[o] = sk; vbuf[o] = sv;
        }
        ssq += sq * sq; ssk += sk * sk;
    }
    atomicAdd(&s_ssq, ssq); atomicAdd(&s_ssk, ssk);
    __syncthreads();
    if (tid == 0) { atomicAdd(&red[0], s_ssq); atomicAdd(&red[1], s_ssk); }
    if (tid < 64) {
        float s = 0.f;
        for (int r2 = 0; r2 < 64; ++r2) s += ks[r2 * 64 + tid];
        atomicAdd(&red[2 + tid], s);
    }
    for (int p = tid; p < 4096; p += 256) {
        int m = p >> 6, d = p & 63;
        float s = 0.f;
        for (int r2 = 0; r2 < 64; ++r2) s += ks[r2 * 64 + m] * vs[r2 * 64 + d];
        atomicAdd(&red[66 + p], s);
    }
}

// ---------------- attention finalize: a -> residual -> LayerNorm ----------------
__global__ void __launch_bounds__(256) attn_fin(
    const float* __restrict__ qb, const float* __restrict__ vb, const float* __restrict__ pin,
    const float* __restrict__ red, const float* __restrict__ lng, const float* __restrict__ lnb,
    float* __restrict__ tout, float* __restrict__ pout, int n)
{
    __shared__ float sKV[4096], sks[64], sinv;
    int tid = threadIdx.x;
    for (int i = tid; i < 4096; i += 256) sKV[i] = red[66 + i];
    if (tid < 64) sks[tid] = red[2 + tid];
    if (tid == 0) sinv = 1.f / (sqrtf(red[0]) * sqrtf(red[1]));
    __syncthreads();
    int row = blockIdx.x * 256 + tid;
    if (row >= n) return;
    float q[64];
#pragma unroll
    for (int i = 0; i < 64; ++i) q[i] = qb[(size_t)row * 64 + i];
    float inv = sinv, dd = 0.f;
#pragma unroll
    for (int i = 0; i < 64; ++i) dd += q[i] * sks[i];
    float rden = 1.f / (dd * inv + (float)Nn);
    const float* vr = vb + (size_t)row * 64;
    const float* pr = pin + (size_t)row * 64;
    float sum = 0.f, sumsq = 0.f;
#pragma unroll 1
    for (int d = 0; d < 64; ++d) {
        float s = 0.f;
#pragma unroll
        for (int m = 0; m < 64; ++m) s += q[m] * sKV[m * 64 + d];
        float tv = 0.5f * ((s * inv + (float)Nn * vr[d]) * rden) + 0.5f * pr[d];
        sum += tv; sumsq += tv * tv;
    }
    float mu = sum * (1.f / 64.f);
    float var = sumsq * (1.f / 64.f) - mu * mu;
    float rs = rsqrtf(var + 1e-5f);
    float* tr = tout + (size_t)row * 64;
    float* pw = pout + (size_t)row * 64;
#pragma unroll 1
    for (int d = 0; d < 64; ++d) {
        float s = 0.f;
#pragma unroll
        for (int m = 0; m < 64; ++m) s += q[m] * sKV[m * 64 + d];
        float tv = 0.5f * ((s * inv + (float)Nn * vr[d]) * rden) + 0.5f * pr[d];
        float y = (tv - mu) * rs * lng[d] + lnb[d];
        tr[d] = y; pw[d] = y;
    }
}

// ---------------- GCN helpers ---------------------------------------------------
__global__ void fill_val(float* p, float v, int n) {
    int i = blockIdx.x * 256 + threadIdx.x; if (i < n) p[i] = v;
}
__global__ void deg_edges(const int* __restrict__ dst, float* deg, int E_) {
    int e = blockIdx.x * 256 + threadIdx.x; if (e < E_) atomicAdd(&deg[dst[e]], 1.0f);
}
__global__ void rsqrt_k(const float* __restrict__ deg, float* __restrict__ dinv, int n) {
    int i = blockIdx.x * 256 + threadIdx.x; if (i < n) dinv[i] = rsqrtf(deg[i]);
}
__global__ void gcn_scatter(const int* __restrict__ src, const int* __restrict__ dst,
                            const float* __restrict__ h, const float* __restrict__ dinv,
                            float* __restrict__ agg, int E_, int C) {
    long gid = (long)blockIdx.x * 256 + threadIdx.x;
    long tot = (long)E_ * C; if (gid >= tot) return;
    int e = (int)(gid / C); int c = (int)(gid - (long)e * C);
    int s = src[e], d = dst[e];
    atomicAdd(&agg[(size_t)d * C + c], h[(size_t)s * C + c] * dinv[s] * dinv[d]);
}
// mode 0: agg + h*dinv^2 + b ; mode 1: relu(BN_1e-5(...))
__global__ void gcn_post(const float* __restrict__ agg, const float* __restrict__ h,
                         const float* __restrict__ dinv, const float* __restrict__ b,
                         const float* __restrict__ bng, const float* __restrict__ bnb,
                         float eps, int mode, float* __restrict__ out, int n, int C) {
    long i = (long)blockIdx.x * 256 + threadIdx.x;
    if (i >= (long)n * C) return;
    int row = (int)(i / C); int c = (int)(i - (long)row * C);
    float dv = dinv[row];
    float s = agg[i] + h[i] * dv * dv + b[c];
    if (mode == 1) {
        s = s * (bng[c] * rsqrtf(1.f + eps)) + bnb[c];
        s = s > 0.f ? s : 0.f;
    }
    out[i] = s;
}
__global__ void axpy2(const float* __restrict__ a, const float* __restrict__ b,
                      float al, float be, float* __restrict__ o, long n) {
    long i = (long)blockIdx.x * 256 + threadIdx.x; if (i < n) o[i] = al * a[i] + be * b[i];
}
__global__ void feat_to_z(const float* __restrict__ f, float* __restrict__ z, int n) {
    int i = blockIdx.x * 256 + threadIdx.x; if (i >= n * 20) return;
    int row = i / 20, c = i - row * 20;
    z[(size_t)row * 28 + c] = f[i];
}
__global__ void mu_to_z(const float* __restrict__ mu, float* __restrict__ z,
                        float* __restrict__ gz, int n) {
    int i = blockIdx.x * 256 + threadIdx.x; if (i >= n * 8) return;
    int row = i >> 3, c = i & 7;
    float v = mu[i];
    z[(size_t)row * 28 + 20 + c] = v;
    gz[i] = v;
}

// ---------------- DEC Student-t soft assignment ---------------------------------
__global__ void qdist(const float* __restrict__ z, const float* __restrict__ clus,
                      float* __restrict__ qo, int n) {
    __shared__ float sc[560];
    for (int i = threadIdx.x; i < 560; i += 256) sc[i] = clus[i];
    __syncthreads();
    int row = blockIdx.x * 256 + threadIdx.x; if (row >= n) return;
    float zr[28];
#pragma unroll
    for (int j = 0; j < 28; ++j) zr[j] = z[(size_t)row * 28 + j];
    float qv[20], s = 0.f;
#pragma unroll
    for (int kc = 0; kc < 20; ++kc) {
        float d2 = 0.f;
#pragma unroll
        for (int j = 0; j < 28; ++j) { float d = zr[j] - sc[kc * 28 + j]; d2 += d * d; }
        float t = powf(1.f + d2 * (1.f / 0.9f), -0.95f);
        qv[kc] = t; s += t;
    }
    float inv = 1.f / s;
#pragma unroll
    for (int kc = 0; kc < 20; ++kc) qo[(size_t)row * 20 + kc] = qv[kc] * inv;
}

// ---------------- host orchestration --------------------------------------------
extern "C" void kernel_launch(void* const* d_in, const int* in_sizes, int n_in,
                              void* d_out, int out_size, void* d_ws, size_t ws_size,
                              hipStream_t stream)
{
    (void)in_sizes; (void)n_in; (void)out_size; (void)ws_size;
    // inputs (setup_inputs dict order, params leaves in insertion order)
    const float* x    = (const float*)d_in[0];
    const float* e0W  = (const float*)d_in[1];
    const float* e0b  = (const float*)d_in[2];
    const float* e0g  = (const float*)d_in[3];
    const float* e0bb = (const float*)d_in[4];
    const float* e1W  = (const float*)d_in[5];
    const float* e1b  = (const float*)d_in[6];
    const float* e1g  = (const float*)d_in[7];
    const float* e1bb = (const float*)d_in[8];
    const float* fc0W = (const float*)d_in[9];
    const float* fc0b = (const float*)d_in[10];
    const float* ln0g = (const float*)d_in[11];
    const float* ln0b = (const float*)d_in[12];
    const float* c1W  = (const float*)d_in[29];
    const float* c1b  = (const float*)d_in[30];
    const float* bng  = (const float*)d_in[31];
    const float* bnb  = (const float*)d_in[32];
    const float* c2W  = (const float*)d_in[33];
    const float* c2b  = (const float*)d_in[34];
    const float* sgW  = (const float*)d_in[35];
    const float* sgb  = (const float*)d_in[36];
    const float* cmW  = (const float*)d_in[37];
    const float* cmb  = (const float*)d_in[38];
    const float* clW  = (const float*)d_in[39];
    const float* clb  = (const float*)d_in[40];
    const float* dWl  = (const float*)d_in[41];
    const float* dbl  = (const float*)d_in[42];
    const float* dgl  = (const float*)d_in[43];
    const float* dbbl = (const float*)d_in[44];
    const float* doW  = (const float*)d_in[45];
    const float* dob  = (const float*)d_in[46];
    const float* clus = (const float*)d_in[47];
    const int*   ei   = (const int*)d_in[48];
    const int* esrc = ei;
    const int* edst = ei + Ee;

    float* out = (float*)d_out;
    float* oZ  = out;                 // [N,28]
    float* oMU = out + 1400000;       // [N,8]
    float* oLV = out + 1800000;       // [N,8]
    float* oDE = out + 2200000;       // [N,3000]
    float* oQ  = out + 152200000;     // [N,20]
    float* oFX = out + 153200000;     // [N,20]
    float* oGZ = out + 154200000;     // [N,8]

    float* ws = (float*)d_ws;
    float* H1    = ws;                  // 1.6M
    float* FEAT  = ws + 1600000;        // 1.0M
    float* T     = ws + 2600000;        // 3.2M
    float* PREV  = ws + 5800000;        // 3.2M
    float* Qb    = ws + 9000000;        // 3.2M  (reused as GCN h2)
    float* Kb    = ws + 12200000;       // 3.2M  (reused as x2)
    float* Vb    = ws + 15400000;       // 3.2M  (reused as conv_x)
    float* G     = ws + 18600000;       // 3.2M
    float* AGG   = ws + 21800000;       // 3.2M
    float* COMB  = ws + 25000000;       // 3.2M
    float* DEG   = ws + 28200000;       // 50k
    float* DINV  = ws + 28250000;       // 50k
    float* HM    = ws + 28300000;       // 400k
    float* HLV   = ws + 28700000;       // 400k
    float* AGGM  = ws + 29100000;       // 400k
    float* AGGLV = ws + 29500000;       // 400k
    float* DECH  = ws + 29900000;       // 1.6M
    float* RED   = ws + 31500000;       // 4162
    unsigned int* WB1 = (unsigned int*)(ws + 31504200); // 48128 u32
    unsigned int* WBO = (unsigned int*)(ws + 31552328); // 48128 u32

    dim3 B(256);
    // weight prep for the two WMMA GEMMs
    prep_w1  <<<188, B, 0, stream>>>(e0W, WB1);
    prep_wout<<<188, B, 0, stream>>>(doW, WBO);

    // encoder
    gemm1_wmma<<<391, B, 0, stream>>>(x, WB1, e0b, e0g, e0bb, H1);
    lin_k<32,20,1><<<196, B, 0, stream>>>(H1, e1W, e1b, e1g, e1bb, 1e-3f, FEAT, oFX, Nn);
    feat_to_z<<<3907, B, 0, stream>>>(FEAT, oZ, Nn);

    // SGFormer transformer branch
    lin_k<20,64,2><<<196, B, 0, stream>>>(FEAT, fc0W, fc0b, ln0g, ln0b, 0.f, T, PREV, Nn);
    for (int L = 0; L < 2; ++L) {
        const float* Wq  = (const float*)d_in[13 + 8 * L];
        const float* bq  = (const float*)d_in[14 + 8 * L];
        const float* Wk  = (const float*)d_in[15 + 8 * L];
        const float* bk  = (const float*)d_in[16 + 8 * L];
        const float* Wv  = (const float*)d_in[17 + 8 * L];
        const float* bv  = (const float*)d_in[18 + 8 * L];
        const float* lng = (const float*)d_in[19 + 8 * L];
        const float* lnb = (const float*)d_in[20 + 8 * L];
        hipMemsetAsync(RED, 0, 4162 * sizeof(float), stream);
        qkv_kernel<<<782, B, 0, stream>>>(T, Wq, bq, Wk, bk, Wv, bv, Qb, Kb, Vb, RED, Nn);
        attn_fin  <<<196, B, 0, stream>>>(Qb, Vb, PREV, RED, lng, lnb, T, PREV, Nn);
    }

    // GCN branch
    fill_val <<<196,  B, 0, stream>>>(DEG, 1.0f, Nn);
    deg_edges<<<1172, B, 0, stream>>>(edst, DEG, Ee);
    rsqrt_k  <<<196,  B, 0, stream>>>(DEG, DINV, Nn);
    lin_k<20,64,0><<<196, B, 0, stream>>>(FEAT, c1W, nullptr, nullptr, nullptr, 0.f, G, nullptr, Nn);
    hipMemsetAsync(AGG, 0, (size_t)Nn * 64 * sizeof(float), stream);
    gcn_scatter<<<75000, B, 0, stream>>>(esrc, edst, G, DINV, AGG, Ee, 64);
    gcn_post   <<<12500, B, 0, stream>>>(AGG, G, DINV, c1b, bng, bnb, 1e-5f, 1, G, Nn, 64);
    lin_k<64,64,0><<<196, B, 0, stream>>>(G, c2W, nullptr, nullptr, nullptr, 0.f, Qb, nullptr, Nn);
    hipMemsetAsync(AGG, 0, (size_t)Nn * 64 * sizeof(float), stream);
    gcn_scatter<<<75000, B, 0, stream>>>(esrc, edst, Qb, DINV, AGG, Ee, 64);
    gcn_post   <<<12500, B, 0, stream>>>(AGG, Qb, DINV, c2b, nullptr, nullptr, 0.f, 0, Kb, Nn, 64);

    // combine + sg_fc
    axpy2<<<12500, B, 0, stream>>>(Kb, T, 0.8f, 0.2f, COMB, (long)Nn * 64);
    lin_k<64,64,0><<<196, B, 0, stream>>>(COMB, sgW, sgb, nullptr, nullptr, 0.f, Vb, nullptr, Nn);

    // VAE heads (GCN convs 64->8)
    lin_k<64,8,0><<<196, B, 0, stream>>>(Vb, cmW, nullptr, nullptr, nullptr, 0.f, HM,  nullptr, Nn);
    lin_k<64,8,0><<<196, B, 0, stream>>>(Vb, clW, nullptr, nullptr, nullptr, 0.f, HLV, nullptr, Nn);
    hipMemsetAsync(AGGM,  0, (size_t)Nn * 8 * sizeof(float), stream);
    hipMemsetAsync(AGGLV, 0, (size_t)Nn * 8 * sizeof(float), stream);
    gcn_scatter<<<9375, B, 0, stream>>>(esrc, edst, HM,  DINV, AGGM,  Ee, 8);
    gcn_scatter<<<9375, B, 0, stream>>>(esrc, edst, HLV, DINV, AGGLV, Ee, 8);
    gcn_post<<<1563, B, 0, stream>>>(AGGM,  HM,  DINV, cmb, nullptr, nullptr, 0.f, 0, oMU, Nn, 8);
    gcn_post<<<1563, B, 0, stream>>>(AGGLV, HLV, DINV, clb, nullptr, nullptr, 0.f, 0, oLV, Nn, 8);
    mu_to_z<<<1563, B, 0, stream>>>(oMU, oZ, oGZ, Nn);

    // decoder
    lin_k<28,32,1><<<196, B, 0, stream>>>(oZ, dWl, dbl, dgl, dbbl, 1e-3f, DECH, nullptr, Nn);
    gemm2_wmma<<<18360, B, 0, stream>>>(DECH, WBO, dob, oDE);

    // DEC soft assignments
    qdist<<<196, B, 0, stream>>>(oZ, clus, oQ, Nn);
}
